// SelfAttention_39316130628151
// MI455X (gfx1250) — matrix operand order repaired and checked by reading, not compiled, not run on hardware
//
#include <hip/hip_runtime.h>

// ---------------------------------------------------------------------------
// SelfAttention for MI455X (gfx1250): bf16 WMMA everywhere, flash-style
// streaming softmax (softmax axis = query axis; rows follow k-projection).
// ---------------------------------------------------------------------------

#define N_TOK 4096
#define D_DIM 1024

typedef __bf16 bf16_t;
typedef __attribute__((ext_vector_type(16))) __bf16 v16bf;
typedef __attribute__((ext_vector_type(8)))  __bf16 v8bf;
typedef __attribute__((ext_vector_type(8)))  float  v8f;

// ---- helpers ---------------------------------------------------------------

__device__ __forceinline__ bf16_t to_bf16(float f) {
  union { float f; unsigned int u; } v; v.f = f;
  unsigned int u = v.u + 0x7fffu + ((v.u >> 16) & 1u);   // round-nearest-even
  union { unsigned short s; bf16_t b; } r; r.s = (unsigned short)(u >> 16);
  return r.b;
}

__device__ __forceinline__ v8f zero_v8f() {
  v8f z;
#pragma unroll
  for (int i = 0; i < 8; ++i) z[i] = 0.0f;
  return z;
}

// A-matrix fragment (16xK tile rows): lane l(0-15) = row l.
// Half 0: K {k0..k0+7, k0+16..k0+23}; half 1: K {k0+8..15, k0+24..31}.
__device__ __forceinline__ v16bf load_a_frag(const bf16_t* base, int ld,
                                             int row0, int k0) {
  int lane = threadIdx.x & 31;
  int half = lane >> 4, l = lane & 15;
  const bf16_t* p = base + (size_t)(row0 + l) * ld + (k0 + half * 8);
  union { v16bf v; v8bf h[2]; } u;
  u.h[0] = *(const v8bf*)p;
  u.h[1] = *(const v8bf*)(p + 16);
  return u.v;
}

// B-matrix fragment (Kx16 tile), fed from a matrix whose ROW n is B's column n.
// Lane l(0-15) = column n0+l; half 0: K {k0..k0+7, k0+16..k0+23}; half 1: K {k0+8..15, k0+24..31}.
__device__ __forceinline__ v16bf load_b_frag(const bf16_t* base, int ld,
                                             int row0, int k0) {
  int lane = threadIdx.x & 31;
  int half = lane >> 4, l = lane & 15;
  const bf16_t* p = base + (size_t)(row0 + l) * ld + (k0 + half * 8);
  union { v16bf v; v8bf h[2]; } u;
  u.h[0] = *(const v8bf*)p;
  u.h[1] = *(const v8bf*)(p + 16);
  return u.v;
}

__device__ __forceinline__ v8f wmma_bf16(v16bf a, v16bf b, v8f c) {
  return __builtin_amdgcn_wmma_f32_16x16x32_bf16(
      /*neg_a=*/false, a, /*neg_b=*/false, b,
      /*c_mod=*/(short)0, c, /*reuse_a=*/false, /*reuse_b=*/false);
}

// ---- kernel 1: cast x to bf16 ---------------------------------------------

__global__ void cast_x_kernel(const float* __restrict__ x,
                              bf16_t* __restrict__ xb, int n) {
  int i = (blockIdx.x * blockDim.x + threadIdx.x) * 4;
  if (i + 3 < n) {
    float4 f = *(const float4*)(x + i);
    union { bf16_t b[4]; uint2 u; } pk;
    pk.b[0] = to_bf16(f.x); pk.b[1] = to_bf16(f.y);
    pk.b[2] = to_bf16(f.z); pk.b[3] = to_bf16(f.w);
    *(uint2*)(xb + i) = pk.u;
  }
}

// ---- kernel 2: cast + transpose the three weight matrices ------------------

__global__ void cast_wT_kernel(const float* __restrict__ Wq,
                               const float* __restrict__ Wk,
                               const float* __restrict__ Wv,
                               bf16_t* __restrict__ Wtq,
                               bf16_t* __restrict__ Wtk,
                               bf16_t* __restrict__ Wtv) {
  int z = blockIdx.y;
  const float* src = (z == 0) ? Wq : (z == 1) ? Wk : Wv;
  bf16_t* dst = (z == 0) ? Wtq : (z == 1) ? Wtk : Wtv;
  int idx = blockIdx.x * blockDim.x + threadIdx.x;       // 0 .. 1M-1
  int n = idx >> 10, k = idx & 1023;
  dst[(size_t)n * D_DIM + k] = to_bf16(src[(size_t)k * D_DIM + n]);
}

// ---- kernel 3: QKV projections --------------------------------------------
// grid = (32, 8, 3); 256 threads. Wave w computes rows [i0,i0+16) x 128 cols.
// z==2 (V) is stored transposed: Vt[d][j].

__global__ void __launch_bounds__(256)
qkv_gemm_kernel(const bf16_t* __restrict__ xb,
                const bf16_t* __restrict__ Wtq,
                const bf16_t* __restrict__ Wtk,
                const bf16_t* __restrict__ Wtv,
                bf16_t* __restrict__ Qp,
                bf16_t* __restrict__ Kp,
                bf16_t* __restrict__ Vt) {
  int wave = threadIdx.x >> 5;
  int lane = threadIdx.x & 31;
  int half = lane >> 4, l = lane & 15;
  int i0 = blockIdx.x * 128 + wave * 16;
  int n0 = blockIdx.y * 128;
  int z  = blockIdx.z;
  const bf16_t* Wt = (z == 0) ? Wtq : (z == 1) ? Wtk : Wtv;

  v8f acc[8];
#pragma unroll
  for (int t = 0; t < 8; ++t) acc[t] = zero_v8f();

  for (int k0 = 0; k0 < D_DIM; k0 += 32) {
    v16bf a = load_a_frag(xb, D_DIM, i0, k0);
#pragma unroll
    for (int t = 0; t < 8; ++t) {
      v16bf b = load_b_frag(Wt, D_DIM, n0 + t * 16, k0);
      acc[t] = wmma_bf16(a, b, acc[t]);
    }
  }

  if (z == 2) {
    // store transposed: row d = n, contiguous 8 m-values per lane.
#pragma unroll
    for (int t = 0; t < 8; ++t) {
      union { bf16_t b[8]; uint4 u; } pk;
#pragma unroll
      for (int r = 0; r < 8; ++r) pk.b[r] = to_bf16(acc[t][r]);
      int n = n0 + t * 16 + l;
      int m = i0 + half * 8;
      *(uint4*)(Vt + (size_t)n * N_TOK + m) = pk.u;
    }
  } else {
    bf16_t* P = (z == 0) ? Qp : Kp;
#pragma unroll
    for (int t = 0; t < 8; ++t)
#pragma unroll
      for (int r = 0; r < 8; ++r)
        P[(size_t)(i0 + r + half * 8) * D_DIM + n0 + t * 16 + l] =
            to_bf16(acc[t][r]);
  }
}

// ---- kernel 4: flash attention over the query axis -------------------------
// Each block owns 16 output rows (rows of Kp). Phase A: 8 waves split the
// 4096 j-range for online (max,sumexp). Phase B: per 128-j chunk, each wave
// makes one P tile in LDS, then all waves do P @ Vt for their 128-d slice.

__global__ void __launch_bounds__(256)
flash_kernel(const bf16_t* __restrict__ Qp,
             const bf16_t* __restrict__ Kp,
             const bf16_t* __restrict__ Vt,
             float* __restrict__ out) {
  __shared__ __align__(16) bf16_t Pchunk[16 * 128];
  __shared__ float sM[8][16];
  __shared__ float sL[8][16];
  __shared__ float fM[16];
  __shared__ float fLinv[16];

  const float scale = 0.03125f;   // 1/sqrt(1024)
  int tid = threadIdx.x;
  int wave = tid >> 5, lane = tid & 31;
  int half = lane >> 4, l = lane & 15;
  int i0 = blockIdx.x * 16;

  // ---------------- Phase A: per-row running max / sumexp -----------------
  float mr[8], lr[8];
#pragma unroll
  for (int r = 0; r < 8; ++r) { mr[r] = -1e30f; lr[r] = 0.0f; }

  int jBase = wave * (N_TOK / 8);
  for (int j0 = jBase; j0 < jBase + N_TOK / 8; j0 += 16) {
    v8f s = zero_v8f();
    for (int k0 = 0; k0 < D_DIM; k0 += 32) {
      v16bf a = load_a_frag(Kp, D_DIM, i0, k0);
      v16bf b = load_b_frag(Qp, D_DIM, j0, k0);
      s = wmma_bf16(a, b, s);
    }
#pragma unroll
    for (int r = 0; r < 8; ++r) {
      float v = s[r] * scale;
      float mx = v;                       // max over the 16-lane half group
      mx = fmaxf(mx, __shfl_xor(mx, 1, 32));
      mx = fmaxf(mx, __shfl_xor(mx, 2, 32));
      mx = fmaxf(mx, __shfl_xor(mx, 4, 32));
      mx = fmaxf(mx, __shfl_xor(mx, 8, 32));
      float mnew = fmaxf(mr[r], mx);
      float e = __expf(v - mnew);
      e += __shfl_xor(e, 1, 32);
      e += __shfl_xor(e, 2, 32);
      e += __shfl_xor(e, 4, 32);
      e += __shfl_xor(e, 8, 32);
      lr[r] = lr[r] * __expf(mr[r] - mnew) + e;
      mr[r] = mnew;
    }
  }
  if (l == 0) {
#pragma unroll
    for (int r = 0; r < 8; ++r) {
      sM[wave][r + half * 8] = mr[r];
      sL[wave][r + half * 8] = lr[r];
    }
  }
  __syncthreads();
  if (tid < 16) {
    float M = -1e30f;
#pragma unroll
    for (int w = 0; w < 8; ++w) M = fmaxf(M, sM[w][tid]);
    float L = 0.0f;
#pragma unroll
    for (int w = 0; w < 8; ++w) L += sL[w][tid] * __expf(sM[w][tid] - M);
    fM[tid] = M;
    fLinv[tid] = 1.0f / L;
  }
  __syncthreads();

  // ---------------- Phase B: O = P @ V, streamed per 128-j chunk ----------
  int d0 = wave * 128;
  v8f o[8];
#pragma unroll
  for (int t = 0; t < 8; ++t) o[t] = zero_v8f();

  for (int c = 0; c < N_TOK / 128; ++c) {
    int j0 = c * 128 + wave * 16;
    v8f s = zero_v8f();
    for (int k0 = 0; k0 < D_DIM; k0 += 32) {
      v16bf a = load_a_frag(Kp, D_DIM, i0, k0);
      v16bf b = load_b_frag(Qp, D_DIM, j0, k0);
      s = wmma_bf16(a, b, s);
    }
#pragma unroll
    for (int r = 0; r < 8; ++r) {
      int row = r + half * 8;
      float p = __expf(s[r] * scale - fM[row]) * fLinv[row];
      Pchunk[row * 128 + wave * 16 + l] = to_bf16(p);
    }
    __syncthreads();
#pragma unroll
    for (int jj = 0; jj < 128; jj += 32) {
      v16bf a = load_a_frag(Pchunk, 128, 0, jj);
#pragma unroll
      for (int t = 0; t < 8; ++t) {
        v16bf b = load_b_frag(Vt, N_TOK, d0 + t * 16, c * 128 + jj);
        o[t] = wmma_bf16(a, b, o[t]);
      }
    }
    __syncthreads();
  }

#pragma unroll
  for (int t = 0; t < 8; ++t)
#pragma unroll
    for (int r = 0; r < 8; ++r)
      out[(size_t)(i0 + r + half * 8) * D_DIM + d0 + t * 16 + l] = o[t][r];
}

// ---------------------------------------------------------------------------

extern "C" void kernel_launch(void* const* d_in, const int* in_sizes, int n_in,
                              void* d_out, int out_size, void* d_ws,
                              size_t ws_size, hipStream_t stream) {
  const float* x  = (const float*)d_in[0];
  const float* Wq = (const float*)d_in[1];
  const float* Wk = (const float*)d_in[2];
  const float* Wv = (const float*)d_in[3];
  float* out = (float*)d_out;

  char* ws = (char*)d_ws;
  const size_t MB = 1024ull * 1024ull;
  bf16_t* xb  = (bf16_t*)(ws + 0 * MB);    // 4096x1024 bf16  (8 MB)
  bf16_t* Wtq = (bf16_t*)(ws + 8 * MB);    // 1024x1024 bf16  (2 MB, transposed)
  bf16_t* Wtk = (bf16_t*)(ws + 10 * MB);
  bf16_t* Wtv = (bf16_t*)(ws + 12 * MB);
  bf16_t* Qp  = (bf16_t*)(ws + 14 * MB);   // 4096x1024 bf16
  bf16_t* Kp  = (bf16_t*)(ws + 22 * MB);   // 4096x1024 bf16
  bf16_t* Vt  = (bf16_t*)(ws + 30 * MB);   // 1024x4096 bf16 (transposed V)

  cast_x_kernel<<<(N_TOK * D_DIM) / (256 * 4), 256, 0, stream>>>(
      x, xb, N_TOK * D_DIM);
  dim3 gw((D_DIM * D_DIM) / 256, 3, 1);
  cast_wT_kernel<<<gw, 256, 0, stream>>>(Wq, Wk, Wv, Wtq, Wtk, Wtv);
  dim3 gg(N_TOK / 128, D_DIM / 128, 3);
  qkv_gemm_kernel<<<gg, 256, 0, stream>>>(xb, Wtq, Wtk, Wtv, Qp, Kp, Vt);
  flash_kernel<<<N_TOK / 16, 256, 0, stream>>>(Qp, Kp, Vt, out);
}
